// Attention_12592844112058
// MI455X (gfx1250) — matrix operand order repaired and checked by reading, not compiled, not run on hardware
//
#include <hip/hip_runtime.h>
#include <hip/hip_bf16.h>

// ---------------------------------------------------------------------------
// Problem constants
// ---------------------------------------------------------------------------
#define B_    4
#define N_    2048
#define D_    1024
#define H_    16
#define DH_   64
#define NKV_  2
#define J_    (N_ + NKV_)     // 2050
#define JP_   2080            // J padded to multiple of 32
#define ROWS_ (B_ * N_)       // 8192

typedef __attribute__((ext_vector_type(16))) __bf16 v16bf;
typedef __attribute__((ext_vector_type(8)))  __bf16 v8bf;
typedef __attribute__((ext_vector_type(8)))  float  v8f;
typedef __attribute__((ext_vector_type(4)))  unsigned int v4u;
typedef __attribute__((ext_vector_type(8)))  int    v8i;
typedef __attribute__((ext_vector_type(4)))  int    v4i;

union FragU { v16bf v; v8bf h[2]; };

#define WMMA_BF16(a, b, c) \
    __builtin_amdgcn_wmma_f32_16x16x32_bf16(false, (a), false, (b), (short)0, (c), false, false)

// ---------------------------------------------------------------------------
// Fragment loaders (per ISA 7.12.2, wave32):
//  A 16x32 (MxK): lane L: m=L&15, half=L>>4; elems 0..7 -> K = half*8+0..7,
//                 elems 8..15 -> K = 16+half*8+0..7  (two contiguous runs)
//  B 32x16 (KxN): lane L: n=L&15, half=L>>4; elems 0..7 -> K = half*8+0..7,
//                 elems 8..15 -> K = 16+half*8+0..7  (two contiguous runs)
// ---------------------------------------------------------------------------
__device__ inline v16bf frag_a(const __bf16* base, int ld, int k0) {
    int lane = threadIdx.x & 31;
    int m = lane & 15, hf = lane >> 4;
    const __bf16* p = base + (size_t)m * ld + k0 + hf * 8;
    FragU f;
    f.h[0] = *(const v8bf*)(p);
    f.h[1] = *(const v8bf*)(p + 16);
    return f.v;
}

// W stored row-major [N,K]; supplies B[k,n] = W[col0+n, k]
__device__ inline v16bf frag_b(const __bf16* W, int col0, int ld, int k0) {
    int lane = threadIdx.x & 31;
    int n = lane & 15, hf = lane >> 4;
    const __bf16* p = W + (size_t)(col0 + n) * ld + k0 + hf * 8;
    FragU f;
    f.h[0] = *(const v8bf*)(p);
    f.h[1] = *(const v8bf*)(p + 16);
    return f.v;
}

// ---------------------------------------------------------------------------
// Tensor Data Mover: 2D tile (bf16) global -> LDS.
// D# packing per CDNA5 ISA:
//   group0: [1:0]=count=1, [63:32]=lds_addr, [120:64]=global_addr, [127:126]=type=2
//   group1: [17:16]=data_size(1 => 2B), [79:48]=tensor_dim0, [111:80]=tensor_dim1,
//           [127:112]=tile_dim0, [143:128]=tile_dim1, [159:144]=tile_dim2,
//           [207:160]=tensor_dim0_stride, [255:208]=tensor_dim1_stride
// dims/strides in data_size units. 2D: descriptor groups 2/3 disabled (zeros).
// ---------------------------------------------------------------------------
__device__ inline void tdm_load_2d_bf16(unsigned int lds_addr, const void* gptr,
                                        unsigned int dim0, unsigned int dim1,
                                        unsigned int tile0, unsigned int tile1,
                                        unsigned long long stride0) {
    unsigned long long ga = (unsigned long long)gptr;
    v4u g0;
    g0[0] = 1u;                                            // count=1, user mode
    g0[1] = lds_addr;                                      // LDS byte address
    g0[2] = (unsigned int)(ga & 0xffffffffull);            // global_addr[31:0]
    g0[3] = (unsigned int)((ga >> 32) & 0x1ffffffull)      // global_addr[56:32]
          | (2u << 30);                                    // type=2 ("image")
    v8i g1;
    g1[0] = (int)(1u << 16);                               // data_size=1 (2 bytes)
    g1[1] = (int)((dim0 & 0xffffu) << 16);                 // tensor_dim0[15:0]
    g1[2] = (int)((dim0 >> 16) | ((dim1 & 0xffffu) << 16));
    g1[3] = (int)((dim1 >> 16) | (tile0 << 16));           // tile_dim0
    g1[4] = (int)(tile1 & 0xffffu);                        // tile_dim1 (tile_dim2=0)
    g1[5] = (int)(stride0 & 0xffffffffull);                // dim0_stride[31:0]
    g1[6] = (int)((stride0 >> 32) & 0xffffull);            // dim0_stride[47:32]
    g1[7] = 0;
    v4i z4 = {};
#if __clang_major__ >= 23
    v8i z8 = {};
    __builtin_amdgcn_tensor_load_to_lds(g0, g1, z4, z4, z8, 0);
#else
    __builtin_amdgcn_tensor_load_to_lds(g0, g1, z4, z4, 0);
#endif
}

// ---------------------------------------------------------------------------
// 1) LayerNorm: per row of 1024, write normed (xn) and raw (xb) as bf16
// ---------------------------------------------------------------------------
__global__ __launch_bounds__(256) void ln_kernel(const float* __restrict__ x,
                                                 const float* __restrict__ gamma,
                                                 __bf16* __restrict__ xn,
                                                 __bf16* __restrict__ xb) {
    __shared__ float ssum[256], ssq[256];
    int tid = threadIdx.x;
    size_t row = blockIdx.x;
    const float* xr = x + row * D_;
    float v[4], lsum = 0.f, lsq = 0.f;
#pragma unroll
    for (int i = 0; i < 4; i++) {
        v[i] = xr[tid + 256 * i];
        lsum += v[i];
        lsq  += v[i] * v[i];
    }
    ssum[tid] = lsum; ssq[tid] = lsq;
    __syncthreads();
    for (int s = 128; s > 0; s >>= 1) {
        if (tid < s) { ssum[tid] += ssum[tid + s]; ssq[tid] += ssq[tid + s]; }
        __syncthreads();
    }
    float mu = ssum[0] * (1.0f / D_);
    float var = ssq[0] * (1.0f / D_) - mu * mu;
    float rs = rsqrtf(var + 1e-5f);
#pragma unroll
    for (int i = 0; i < 4; i++) {
        int c = tid + 256 * i;
        xn[row * D_ + c] = (__bf16)((v[i] - mu) * rs * gamma[c]);
        xb[row * D_ + c] = (__bf16)v[i];
    }
}

// ---------------------------------------------------------------------------
// 2) f32 -> bf16 conversion (weights)
// ---------------------------------------------------------------------------
__global__ __launch_bounds__(256) void f2bf_kernel(const float* __restrict__ in,
                                                   __bf16* __restrict__ out, int n) {
    int i = blockIdx.x * 256 + threadIdx.x;
    if (i < n) out[i] = (__bf16)in[i];
}

// ---------------------------------------------------------------------------
// 3) Generic NT GEMM: C[M,N] = scale * (A[M,K] @ W[N,K]^T)
//    one wave per block; wave tile = 32 rows x 64 cols; K stepped by 32.
//    All 6 fragments of a K-step preloaded into distinct regs so the 8 WMMAs
//    issue behind staggered loadcnt waits instead of wait-0 per WMMA.
// ---------------------------------------------------------------------------
__global__ __launch_bounds__(32) void gemm_nt_bf16(const __bf16* __restrict__ A,
                                                   const __bf16* __restrict__ W,
                                                   int M, int N, int K, float scale,
                                                   __bf16* __restrict__ Obf,
                                                   float* __restrict__ Of) {
    int lane = threadIdx.x & 31;
    int row0 = blockIdx.x * 32;
    int col0 = blockIdx.y * 64;
    const __bf16* A0 = A + (size_t)row0 * K;
    const __bf16* A1 = A + (size_t)(row0 + 16) * K;

    v8f acc0[4] = {};
    v8f acc1[4] = {};
    for (int k0 = 0; k0 < K; k0 += 32) {
        __builtin_prefetch(A0 + k0 + 256, 0, 3);           // global_prefetch_b8
        v16bf a0 = frag_a(A0, K, k0);
        v16bf a1 = frag_a(A1, K, k0);
        v16bf b0 = frag_b(W, col0,      K, k0);
        v16bf b1 = frag_b(W, col0 + 16, K, k0);
        v16bf b2 = frag_b(W, col0 + 32, K, k0);
        v16bf b3 = frag_b(W, col0 + 48, K, k0);
        acc0[0] = WMMA_BF16(a0, b0, acc0[0]);
        acc0[1] = WMMA_BF16(a0, b1, acc0[1]);
        acc0[2] = WMMA_BF16(a0, b2, acc0[2]);
        acc0[3] = WMMA_BF16(a0, b3, acc0[3]);
        acc1[0] = WMMA_BF16(a1, b0, acc1[0]);
        acc1[1] = WMMA_BF16(a1, b1, acc1[1]);
        acc1[2] = WMMA_BF16(a1, b2, acc1[2]);
        acc1[3] = WMMA_BF16(a1, b3, acc1[3]);
    }
    int n = lane & 15, hf = lane >> 4;
#pragma unroll
    for (int t = 0; t < 4; t++) {
#pragma unroll
        for (int r = 0; r < 8; r++) {
            size_t col = (size_t)col0 + t * 16 + n;
            size_t rA = (size_t)row0 + hf * 8 + r;
            size_t rB = rA + 16;
            float vA = acc0[t][r] * scale;
            float vB = acc1[t][r] * scale;
            if (Of) {
                Of[rA * N + col] = vA;
                Of[rB * N + col] = vB;
            } else {
                Obf[rA * N + col] = (__bf16)vA;
                Obf[rB * N + col] = (__bf16)vB;
            }
        }
    }
}

// ---------------------------------------------------------------------------
// 4) Scatter KV: prepend null tokens, zero-pad J->JP,
//    K -> [B][JP][64] row-major, V -> [B][64][JP] transposed.
// ---------------------------------------------------------------------------
__global__ __launch_bounds__(256) void scatter_kv_kernel(const __bf16* __restrict__ kv,
                                                         const float* __restrict__ nullkv,
                                                         __bf16* __restrict__ kbuf,
                                                         __bf16* __restrict__ vT) {
    int e = blockIdx.x * 256 + threadIdx.x;     // B*JP*64 elements, exact grid
    int b = e / (JP_ * DH_);
    int rem = e % (JP_ * DH_);
    int j = rem / DH_;
    int d = rem % DH_;
    __bf16 kk, vv;
    if (j < NKV_) {
        kk = (__bf16)nullkv[j * DH_ + d];                 // null_kv[0][j][d]
        vv = (__bf16)nullkv[NKV_ * DH_ + j * DH_ + d];    // null_kv[1][j][d]
    } else if (j < J_) {
        size_t r = (size_t)b * N_ + (j - NKV_);
        kk = kv[r * (2 * DH_) + d];
        vv = kv[r * (2 * DH_) + DH_ + d];
    } else {
        kk = (__bf16)0.0f;
        vv = (__bf16)0.0f;
    }
    kbuf[((size_t)b * JP_ + j) * DH_ + d] = kk;
    vT[((size_t)b * DH_ + d) * JP_ + j]   = vv;
}

// ---------------------------------------------------------------------------
// 5) Flash attention: one wave per (b, h, 16-row q tile).
//    K/V tiles staged into LDS by the Tensor Data Mover (tensor_load_to_lds,
//    s_wait_tensorcnt); online softmax with 16-lane shfl_xor row reductions;
//    P transposed C-layout -> A-layout through a 1KB LDS buffer.
// ---------------------------------------------------------------------------
__global__ __launch_bounds__(32) void flash_kernel(const __bf16* __restrict__ q,
                                                   const __bf16* __restrict__ kbuf,
                                                   const __bf16* __restrict__ vT,
                                                   const unsigned char* __restrict__ mask,
                                                   __bf16* __restrict__ ao) {
    int lane = threadIdx.x & 31;
    int n = lane & 15, hf = lane >> 4;
    int qt = blockIdx.x, h = blockIdx.y, b = blockIdx.z;
    size_t row0 = (size_t)b * N_ + (size_t)qt * 16;

    __shared__ __bf16 k_lds[32][64];   // K tile: 32 j-rows x 64 (4 KB)
    __shared__ __bf16 v_lds[64][32];   // V^T tile: 64 dh-rows x 32 j (4 KB)
    __shared__ __bf16 p_lds[16][32];   // P transpose staging (1 KB)

    unsigned int k_off = (unsigned int)(size_t)&k_lds[0][0];  // LDS byte addr
    unsigned int v_off = (unsigned int)(size_t)&v_lds[0][0];

    const __bf16* qbase = q + row0 * (H_ * DH_) + h * DH_;
    v16bf qa0 = frag_a(qbase, H_ * DH_, 0);    // q tile 16x64 (scale pre-folded)
    v16bf qa1 = frag_a(qbase, H_ * DH_, 32);

    const __bf16* kb_base = kbuf + (size_t)b * JP_ * DH_;
    const __bf16* vt_base = vT + (size_t)b * DH_ * JP_;
    const unsigned char* mrow = mask + (size_t)b * N_;

    v8f acc[4] = {};
    float mrun[8], lrun[8];
#pragma unroll
    for (int r = 0; r < 8; r++) { mrun[r] = -1e30f; lrun[r] = 0.0f; }

    for (int j0 = 0; j0 < JP_; j0 += 32) {
        // drain previous iteration's LDS reads before TDM overwrites the tiles
        asm volatile("s_wait_dscnt 0x0" ::: "memory");
        // K tile: rows j0..j0+31 of [JP,64];  V tile: 64 rows x cols j0..j0+31 of [64,JP]
        tdm_load_2d_bf16(k_off, kb_base + (size_t)j0 * DH_, DH_, JP_, DH_, 32, DH_);
        tdm_load_2d_bf16(v_off, vt_base + j0, JP_, DH_, 32, DH_, JP_);
        __builtin_amdgcn_s_wait_tensorcnt(0);
        asm volatile("" ::: "memory");   // fence: LDS tiles were written by TDM

        // sim[16, 32] = q[16,64] @ k[j0:j0+32, 64]^T  (two 16x16 j tiles)
        v8f s0 = {}, s1 = {};
        {
            v16bf kb00 = frag_b(&k_lds[0][0], 0,  DH_, 0);
            v16bf kb01 = frag_b(&k_lds[0][0], 0,  DH_, 32);
            v16bf kb10 = frag_b(&k_lds[0][0], 16, DH_, 0);
            v16bf kb11 = frag_b(&k_lds[0][0], 16, DH_, 32);
            s0 = WMMA_BF16(qa0, kb00, s0);
            s0 = WMMA_BF16(qa1, kb01, s0);
            s1 = WMMA_BF16(qa0, kb10, s1);
            s1 = WMMA_BF16(qa1, kb11, s1);
        }
        // masking (null tokens always valid; padded / masked-out j -> -inf)
        int jg0 = j0 + n, jg1 = j0 + 16 + n;
        bool ok0 = (jg0 < NKV_) || (jg0 < J_ && mrow[jg0 - NKV_] != 0);
        bool ok1 = (jg1 < NKV_) || (jg1 < J_ && mrow[jg1 - NKV_] != 0);

        // online softmax: row m = hf*8 + r lives across one 16-lane half
#pragma unroll
        for (int r = 0; r < 8; r++) {
            float a0 = ok0 ? s0[r] : -1e30f;
            float a1 = ok1 ? s1[r] : -1e30f;
            float mx = fmaxf(a0, a1);
            mx = fmaxf(mx, __shfl_xor(mx, 1));
            mx = fmaxf(mx, __shfl_xor(mx, 2));
            mx = fmaxf(mx, __shfl_xor(mx, 4));
            mx = fmaxf(mx, __shfl_xor(mx, 8));
            float mi    = fmaxf(mrun[r], mx);
            float alpha = __expf(mrun[r] - mi);
            mrun[r] = mi;
            float p0 = __expf(a0 - mi);
            float p1 = __expf(a1 - mi);
            float ls = p0 + p1;
            ls += __shfl_xor(ls, 1);
            ls += __shfl_xor(ls, 2);
            ls += __shfl_xor(ls, 4);
            ls += __shfl_xor(ls, 8);
            lrun[r] = lrun[r] * alpha + ls;
#pragma unroll
            for (int t = 0; t < 4; t++) acc[t][r] = acc[t][r] * alpha;
            // stash P in C-layout into LDS
            p_lds[hf * 8 + r][n]      = (__bf16)p0;
            p_lds[hf * 8 + r][16 + n] = (__bf16)p1;
        }
        __syncthreads();

        // reload P as A-layout fragment (16x32 over j)
        v16bf pa;
        {
            const __bf16* pp = &p_lds[n][hf * 8];
            FragU f;
            f.h[0] = *(const v8bf*)(pp);
            f.h[1] = *(const v8bf*)(pp + 16);
            pa = f.v;
        }
        // out[16,64] += P[16,32] @ v[j0:j0+32, 64]  (B frags from LDS V^T tile)
#pragma unroll
        for (int t = 0; t < 4; t++) {
            v16bf vb = frag_b(&v_lds[0][0], t * 16, 32, 0);
            acc[t] = WMMA_BF16(pa, vb, acc[t]);
        }
        __syncthreads();
    }

    // epilogue: normalize by row sums, write bf16 [b, i, h*64 + c]
#pragma unroll
    for (int r = 0; r < 8; r++) {
        float inv = 1.0f / lrun[r];
        size_t row = row0 + hf * 8 + r;
#pragma unroll
        for (int t = 0; t < 4; t++) {
            ao[row * (H_ * DH_) + h * DH_ + t * 16 + n] = (__bf16)(acc[t][r] * inv);
        }
    }
}

// ---------------------------------------------------------------------------
// Launch
// ---------------------------------------------------------------------------
extern "C" void kernel_launch(void* const* d_in, const int* in_sizes, int n_in,
                              void* d_out, int out_size, void* d_ws, size_t ws_size,
                              hipStream_t stream) {
    const float*         x      = (const float*)d_in[0];
    const unsigned char* mask   = (const unsigned char*)d_in[1];
    const float*         gamma  = (const float*)d_in[2];
    const float*         nullkv = (const float*)d_in[3];
    const float*         Wq     = (const float*)d_in[4];
    const float*         Wkv    = (const float*)d_in[5];
    const float*         Wo     = (const float*)d_in[6];
    float*               out    = (float*)d_out;

    char* cur = (char*)d_ws;
    auto alloc = [&](size_t bytes) -> char* {
        char* p = cur;
        cur += (bytes + 255) & ~(size_t)255;
        return p;
    };
    __bf16* xn_bf  = (__bf16*)alloc((size_t)ROWS_ * D_ * 2);          // 16 MB
    __bf16* x_bf   = (__bf16*)alloc((size_t)ROWS_ * D_ * 2);          // 16 MB
    __bf16* wq_bf  = (__bf16*)alloc((size_t)D_ * D_ * 2);             //  2 MB
    __bf16* wkv_bf = (__bf16*)alloc((size_t)(2 * DH_) * D_ * 2);      // .25 MB
    __bf16* wo_bf  = (__bf16*)alloc((size_t)D_ * D_ * 2);             //  2 MB
    __bf16* q_bf   = (__bf16*)alloc((size_t)ROWS_ * D_ * 2);          // 16 MB
    __bf16* kv_bf  = (__bf16*)alloc((size_t)ROWS_ * 2 * DH_ * 2);     //  2 MB
    __bf16* k_bf   = (__bf16*)alloc((size_t)B_ * JP_ * DH_ * 2);      //  1 MB
    __bf16* vT_bf  = (__bf16*)alloc((size_t)B_ * DH_ * JP_ * 2);      //  1 MB
    __bf16* ao_bf  = (__bf16*)alloc((size_t)ROWS_ * D_ * 2);          // 16 MB

    // 1) LayerNorm + f32->bf16 staging of x
    ln_kernel<<<ROWS_, 256, 0, stream>>>(x, gamma, xn_bf, x_bf);

    // 2) weight conversions
    f2bf_kernel<<<(D_ * D_) / 256, 256, 0, stream>>>(Wq, wq_bf, D_ * D_);
    f2bf_kernel<<<(2 * DH_ * D_) / 256, 256, 0, stream>>>(Wkv, wkv_bf, 2 * DH_ * D_);
    f2bf_kernel<<<(D_ * D_) / 256, 256, 0, stream>>>(Wo, wo_bf, D_ * D_);

    // 3) q = xn @ Wq^T, scale 1/sqrt(64) folded in
    gemm_nt_bf16<<<dim3(ROWS_ / 32, D_ / 64), 32, 0, stream>>>(
        xn_bf, wq_bf, ROWS_, D_, D_, 0.125f, q_bf, nullptr);

    // 4) kv = x @ Wkv^T
    gemm_nt_bf16<<<dim3(ROWS_ / 32, (2 * DH_) / 64), 32, 0, stream>>>(
        x_bf, wkv_bf, ROWS_, 2 * DH_, D_, 1.0f, kv_bf, nullptr);

    // 5) assemble K [B][JP][64] and V^T [B][64][JP] (+null tokens, zero pad)
    scatter_kv_kernel<<<(B_ * JP_ * DH_) / 256, 256, 0, stream>>>(
        kv_bf, nullkv, k_bf, vT_bf);

    // 6) attention (TDM-staged K/V)
    flash_kernel<<<dim3(N_ / 16, H_, B_), 32, 0, stream>>>(
        q_bf, k_bf, vT_bf, mask, ao_bf);

    // 7) out = ao @ Wo^T  (fp32 result)
    gemm_nt_bf16<<<dim3(ROWS_ / 32, D_ / 64), 32, 0, stream>>>(
        ao_bf, wo_bf, ROWS_, D_, D_, 1.0f, nullptr, out);
}
